// KNF_CNN_GLOBAL_19310172963082
// MI455X (gfx1250) — matrix operand order repaired and checked by reading, not compiled, not run on hardware
//
#include <hip/hip_runtime.h>
#include <hip/hip_bf16.h>
#include <cstdint>

// ---------------------------------------------------------------------------
// KNF forward pass for MI455X (gfx1250, wave32, WMMA + TDM).
//  * encode/decode chains are per-(b,n) independent (all convs are 1xK along T)
//    -> fully fused per-row kernels: activations live in LDS only.
//  * channel-mixing convs = implicit GEMM via v_wmma_f32_16x16x32_f16.
//    A-fragment loads kept inside a ROLLED kc loop (loop-variant addresses =>
//    no LICM hoist => no scratch spills).  LDS tiles padded +-2 t-rows with
//    zeros => unconditional ds_load_b128 B-fragment loads (no EXEC juggling).
//  * rollout: K (360x360 f16) staged into LDS by the Tensor Data Mover
//    (tensor_load_to_lds + s_wait_tensorcnt), serial matvec iterations.
//  * classifier convs: generic global WMMA implicit-GEMM (Cin=360/128 -> 128).
// ---------------------------------------------------------------------------

typedef __attribute__((ext_vector_type(16))) _Float16 v16h;
typedef __attribute__((ext_vector_type(8)))  _Float16 v8h;
typedef __attribute__((ext_vector_type(8)))  float    v8f;
typedef unsigned int u32x4 __attribute__((ext_vector_type(4)));
typedef int          i32x4 __attribute__((ext_vector_type(4)));
typedef int          i32x8 __attribute__((ext_vector_type(8)));

constexpr int Bc  = 8;
constexpr int Tc  = 384;
constexpr int TYc = 96;
constexpr int Nc  = 360;
constexpr int Hc  = 64;
constexpr int CHc = 128;
constexpr int NCc = 21;

constexpr int PADT = 2;                 // zero-padded t rows on each side
constexpr int TROWS = Tc + 2 * PADT;    // padded tile rows (worst case T)

// d_out layout (float32, reference tuple order, flattened)
constexpr size_t O_XREC   = 0;
constexpr size_t SZ_XREC  = (size_t)Bc * Nc * Tc;          // 1105920
constexpr size_t O_LBREC  = O_XREC + SZ_XREC;
constexpr size_t SZ_LBREC = (size_t)Bc * Nc * (Tc - 1);    // 1103040
constexpr size_t O_LAREC  = O_LBREC + SZ_LBREC;
constexpr size_t SZ_LAREC = (size_t)Bc * Nc * TYc;         // 276480
constexpr size_t O_SCORES = O_LAREC + SZ_LAREC;
constexpr size_t O_SCAL   = O_SCORES + (size_t)Bc * NCc;   // 5 scalars follow

__device__ __forceinline__ v8f wmma16x16x32(v16h a, v16h b, v8f c) {
    // D = A(16x32 f16) * B(32x16 f16) + C(16x16 f32)
    return __builtin_amdgcn_wmma_f32_16x16x32_f16(
        false, a, false, b, (short)0, c, false, false);
}

// ---------------------------------------------------------------------------
// Pack a conv weight (Cout, Cin, 3) [optionally transposed (Cin, Cout, 3)]
// into WMMA A-fragment order: Apack[((mt*KC + kc)*32 + lane)*16 + slot].
// K ordering: k = j*Cin + ci (tap-major).  A-layout used here (contiguous k, same as B):
//   lane L holds M = mt*16 + (L&15); slot s: K = 16*(L>>4) + s.
// ---------------------------------------------------------------------------
__global__ void pack_a_kernel(const float* __restrict__ W,
                              _Float16* __restrict__ Apack,
                              int Cout, int Cin, int KC, int transpose, int total) {
    int idx = blockIdx.x * blockDim.x + threadIdx.x;
    if (idx >= total) return;
    int slot = idx & 15;
    int lane = (idx >> 4) & 31;
    int chunk = idx >> 9;
    int kc = chunk % KC;
    int mt = chunk / KC;
    int m = mt * 16 + (lane & 15);
    int v = slot >> 1, p = slot & 1;
    int K = ((lane >> 4) << 4) + (v << 1) + p;
    int k = kc * 32 + K;
    int KK = Cin * 3;
    float val = 0.f;
    if (k < KK && m < Cout) {
        int j  = (k >= 2 * Cin) ? 2 : ((k >= Cin) ? 1 : 0);
        int ci = k - j * Cin;
        val = transpose ? W[(ci * Cout + m) * 3 + j]
                        : W[(m * Cin + ci) * 3 + j];
    }
    Apack[idx] = (_Float16)val;
}

// ---------------------------------------------------------------------------
// One 64->64 conv layer (kernel 3 along T) computed in LDS via WMMA.
// src/dst point at row t=0 of a tile padded with PADT zeroed rows both sides;
// layout [t*64 + ch] (f16).  Tap offset for tap j is offA + offB*j
// (encode: -1,+1 ; decode/convT: 0,-1).  Per 16-wide t tile:
// M=64 (4 mt tiles), K=192 (6 chunks of 32), N=16.
// kc loop deliberately NOT unrolled: keeps A-fragment loads loop-variant so
// they are not hoisted/spilled; pressure stays ~80 VGPRs.
// ---------------------------------------------------------------------------
__device__ void conv64_wmma_lds(const _Float16* src, _Float16* dst,
                                const _Float16* __restrict__ Apack,
                                const float* __restrict__ bias,
                                int relu, int offA, int offB, int T) {
    const int lane   = threadIdx.x & 31;
    const int wave   = threadIdx.x >> 5;
    const int nwaves = blockDim.x >> 5;
    const int ntiles = (T + 15) >> 4;
    for (int tile = wave; tile < ntiles; tile += nwaves) {
        const int t0 = tile << 4;
        v8f c0 = {}, c1 = {}, c2 = {}, c3 = {};
#pragma unroll 1
        for (int kc = 0; kc < 6; ++kc) {
            const int j = kc >> 1;
            const int cibase = (kc & 1) << 5;
            const int tin = t0 + (lane & 15) + offA + offB * j;   // in [-2, T+1]
            v16h bf = *(const v16h*)(src + tin * 64 + cibase + ((lane >> 4) << 4));
            v16h a0 = *(const v16h*)(Apack + (((0 * 6 + kc) * 32) + lane) * 16);
            v16h a1 = *(const v16h*)(Apack + (((1 * 6 + kc) * 32) + lane) * 16);
            v16h a2 = *(const v16h*)(Apack + (((2 * 6 + kc) * 32) + lane) * 16);
            v16h a3 = *(const v16h*)(Apack + (((3 * 6 + kc) * 32) + lane) * 16);
            c0 = wmma16x16x32(a0, bf, c0);
            c1 = wmma16x16x32(a1, bf, c1);
            c2 = wmma16x16x32(a2, bf, c2);
            c3 = wmma16x16x32(a3, bf, c3);
        }
        const int t = t0 + (lane & 15);
        if (t < T) {
            const int mrow = (lane >> 4) << 3;
#pragma unroll
            for (int r = 0; r < 8; ++r) {
                int m = mrow + r;
                float v0 = c0[r] + bias[m];
                float v1 = c1[r] + bias[16 + m];
                float v2 = c2[r] + bias[32 + m];
                float v3 = c3[r] + bias[48 + m];
                if (relu) {
                    v0 = v0 > 0.f ? v0 : 0.f;  v1 = v1 > 0.f ? v1 : 0.f;
                    v2 = v2 > 0.f ? v2 : 0.f;  v3 = v3 > 0.f ? v3 : 0.f;
                }
                dst[t * 64 + m]      = (_Float16)v0;
                dst[t * 64 + 16 + m] = (_Float16)v1;
                dst[t * 64 + 32 + m] = (_Float16)v2;
                dst[t * 64 + 48 + m] = (_Float16)v3;
            }
        }
    }
}

// Zero the PADT pad rows (t in {-2,-1,T,T+1}) of both tile buffers.
__device__ __forceinline__ void zero_pads(_Float16* bufA, _Float16* bufB, int T) {
    for (int i = threadIdx.x; i < 4 * 64; i += blockDim.x) {
        int r = i >> 6, c = i & 63;
        int t = (r < 2) ? (r - 2) : (T + r - 2);
        bufA[t * 64 + c] = (_Float16)0.f;
        bufB[t * 64 + c] = (_Float16)0.f;
    }
}

// ---------------------------------------------------------------------------
// Fused encoder: x row -> conv1(1->64,relu) -> conv2(relu) -> conv3 -> proj
// One block per (b,n).  All intermediates in LDS.
// ---------------------------------------------------------------------------
__global__ void encode_fused_kernel(const float* __restrict__ x,       // (B,T,N)
                                    const float* __restrict__ w0,      // (64,3)
                                    const float* __restrict__ b0,
                                    const _Float16* __restrict__ Ap1,
                                    const float* __restrict__ b1,
                                    const _Float16* __restrict__ Ap2,
                                    const float* __restrict__ b2,
                                    const float* __restrict__ pw,
                                    const float* __restrict__ pb,
                                    float* __restrict__ z32,           // (B,N,T)
                                    _Float16* __restrict__ z16) {
    extern __shared__ char smem[];
    _Float16* bufA = (_Float16*)smem + PADT * 64;                  // row t=0
    _Float16* bufB = (_Float16*)smem + (TROWS + PADT) * 64;
    float*    xrow = (float*)((_Float16*)smem + 2 * TROWS * 64);
    const int bn = blockIdx.x;
    const int b = bn / Nc, n = bn % Nc;

    zero_pads(bufA, bufB, Tc);
    for (int t = threadIdx.x; t < Tc; t += blockDim.x)
        xrow[t] = x[((size_t)b * Tc + t) * Nc + n];
    __syncthreads();

    for (int idx = threadIdx.x; idx < Tc * 64; idx += blockDim.x) {
        int t = idx >> 6, co = idx & 63;
        float acc = b0[co];
#pragma unroll
        for (int j = 0; j < 3; ++j) {
            int ti = t + j - 1;
            if (ti >= 0 && ti < Tc) acc += xrow[ti] * w0[co * 3 + j];
        }
        bufA[t * 64 + co] = (_Float16)(acc > 0.f ? acc : 0.f);
    }
    __syncthreads();
    conv64_wmma_lds(bufA, bufB, Ap1, b1, 1, -1, 1, Tc);
    __syncthreads();
    conv64_wmma_lds(bufB, bufA, Ap2, b2, 0, -1, 1, Tc);
    __syncthreads();

    const float pbv = pb[0];
    for (int t = threadIdx.x; t < Tc; t += blockDim.x) {
        float acc = pbv;
#pragma unroll
        for (int c = 0; c < 64; ++c) acc += (float)bufA[t * 64 + c] * pw[c];
        size_t o = (size_t)bn * Tc + t;
        z32[o] = acc;
        z16[o] = (_Float16)acc;
    }
}

// ---------------------------------------------------------------------------
// Fused decoder: latent row -> convT0(1->64,relu) -> convT1(relu) -> convT2
// -> proj -> recon (f32, straight to d_out).  ConvT_full(+trunc) == causal
// correlation: out[t] = sum_j in[t-j] * Weff[o][i][j], Weff = transpose(w).
// ---------------------------------------------------------------------------
__global__ void decode_fused_kernel(const _Float16* __restrict__ zin,  // (B*N, T)
                                    const float* __restrict__ wt0,     // (64,3)
                                    const float* __restrict__ bt0,
                                    const _Float16* __restrict__ Ap1,
                                    const float* __restrict__ b1,
                                    const _Float16* __restrict__ Ap2,
                                    const float* __restrict__ b2,
                                    const float* __restrict__ pw,
                                    const float* __restrict__ pb,
                                    float* __restrict__ out, int T) {
    extern __shared__ char smem[];
    _Float16* bufA = (_Float16*)smem + PADT * 64;                  // row t=0
    _Float16* bufB = (_Float16*)smem + (TROWS + PADT) * 64;
    const int bn = blockIdx.x;
    const _Float16* zr = zin + (size_t)bn * T;

    zero_pads(bufA, bufB, T);
    __syncthreads();

    for (int idx = threadIdx.x; idx < T * 64; idx += blockDim.x) {
        int t = idx >> 6, co = idx & 63;
        float acc = bt0[co];
#pragma unroll
        for (int j = 0; j < 3; ++j) {
            int ti = t - j;
            if (ti >= 0) acc += (float)zr[ti] * wt0[co * 3 + j];
        }
        bufA[t * 64 + co] = (_Float16)(acc > 0.f ? acc : 0.f);
    }
    __syncthreads();
    conv64_wmma_lds(bufA, bufB, Ap1, b1, 1, 0, -1, T);
    __syncthreads();
    conv64_wmma_lds(bufB, bufA, Ap2, b2, 0, 0, -1, T);
    __syncthreads();

    const float pbv = pb[0];
    for (int t = threadIdx.x; t < T; t += blockDim.x) {
        float acc = pbv;
#pragma unroll
        for (int c = 0; c < 64; ++c) acc += (float)bufA[t * 64 + c] * pw[c];
        out[(size_t)bn * T + t] = acc;
    }
}

// ---------------------------------------------------------------------------
// Generic conv1d (kernel 3, same-pad) as global implicit GEMM, Cout = 128.
// X: (B, Cin, T) f16.  One wave per (b, 16-wide t tile); 8 M-tiles per wave.
// A-fragment addresses are kc-dependent -> stay in loop, no spills.
// ---------------------------------------------------------------------------
__global__ void conv1d_wmma_global_kernel(const _Float16* __restrict__ X,
                                          const _Float16* __restrict__ Apack,
                                          const float* __restrict__ bias,
                                          _Float16* __restrict__ out,
                                          int Cin, int T, int relu) {
    const int lane = threadIdx.x & 31;
    const int gw   = (blockIdx.x * blockDim.x + threadIdx.x) >> 5;
    const int ntiles = (T + 15) >> 4;
    if (gw >= Bc * ntiles) return;
    const int b  = gw / ntiles;
    const int t0 = (gw % ntiles) << 4;
    const int KK = Cin * 3;
    const int KC = (KK + 31) >> 5;

    v8f c[8];
#pragma unroll
    for (int i = 0; i < 8; ++i) c[i] = (v8f){};

    const _Float16* Xb = X + (size_t)b * Cin * T;
    const int tcol = t0 + (lane & 15);
#pragma unroll 1
    for (int kc = 0; kc < KC; ++kc) {
        // hint: pull next chunk's A-fragments toward the caches (global_prefetch_b8)
        if (kc + 1 < KC)
            __builtin_prefetch(Apack + (((size_t)(kc + 1)) * 32 + lane) * 16, 0, 0);
        const int kb = kc * 32 + ((lane >> 4) << 4);
        v16h bf = {};
#pragma unroll
        for (int s = 0; s < 16; ++s) {
            int k = kb + s;
            float v = 0.f;
            if (k < KK) {
                int j  = (k >= 2 * Cin) ? 2 : ((k >= Cin) ? 1 : 0);
                int ci = k - j * Cin;
                int tin = tcol + j - 1;
                if (tin >= 0 && tin < T) v = (float)Xb[(size_t)ci * T + tin];
            }
            bf[s] = (_Float16)v;
        }
#pragma unroll
        for (int mt = 0; mt < 8; ++mt) {
            v16h a = *(const v16h*)(Apack + (((size_t)mt * KC + kc) * 32 + lane) * 16);
            c[mt] = wmma16x16x32(a, bf, c[mt]);
        }
    }
    const int t = t0 + (lane & 15);
    if (t < T) {
        const int mrow = (lane >> 4) << 3;
#pragma unroll
        for (int mt = 0; mt < 8; ++mt) {
#pragma unroll
            for (int r = 0; r < 8; ++r) {
                int co = mt * 16 + mrow + r;
                float v = c[mt][r] + bias[co];
                if (relu && v < 0.f) v = 0.f;
                out[((size_t)b * CHc + co) * T + t] = (_Float16)v;
            }
        }
    }
}

__global__ void meanpool_kernel(const _Float16* __restrict__ h, float* __restrict__ pooled) {
    int idx = blockIdx.x * blockDim.x + threadIdx.x;
    if (idx >= Bc * CHc) return;
    const _Float16* row = h + (size_t)idx * Tc;
    float s = 0.f;
    for (int t = 0; t < Tc; ++t) s += (float)row[t];
    pooled[idx] = s / (float)Tc;
}

// dense + softmax + prob + cls_loss (single small block)
__global__ void head_kernel(const float* __restrict__ pooled,
                            const float* __restrict__ dw, const float* __restrict__ db,
                            const int* __restrict__ labels,
                            float* __restrict__ scores_out,
                            float* __restrict__ prob,
                            float* __restrict__ loss_out) {
    __shared__ float sc[Bc][NCc];
    __shared__ float lossb[Bc];
    int tid = threadIdx.x;
    if (tid < Bc * NCc) {
        int b = tid / NCc, c = tid % NCc;
        float a = db[c];
        for (int k = 0; k < CHc; ++k) a += pooled[b * CHc + k] * dw[c * CHc + k];
        sc[b][c] = a;
        scores_out[tid] = a;
    }
    __syncthreads();
    if (tid < Bc) {
        int b = tid;
        float m = sc[b][0];
        for (int c = 1; c < NCc; ++c) m = fmaxf(m, sc[b][c]);
        float se = 0.f;
        for (int c = 0; c < NCc; ++c) se += __expf(sc[b][c] - m);
        for (int c = 0; c < NCc; ++c) prob[b * NCc + c] = __expf(sc[b][c] - m) / se;
        float lz = __logf(se) + m;
        int lbl = labels[b];
        lossb[b] = -(sc[b][lbl] - lz) / (float)Bc;
    }
    __syncthreads();
    if (tid == 0) {
        float s = 0.f;
        for (int b = 0; b < Bc; ++b) s += lossb[b];
        loss_out[0] = s;
    }
}

// K[b] = sum_c prob[b,c] * attention[c]   (f16 output for LDS rollout)
__global__ void kbuild_kernel(const float* __restrict__ prob,
                              const float* __restrict__ att,
                              _Float16* __restrict__ K16) {
    const int NN = Nc * Nc;
    int idx = blockIdx.x * blockDim.x + threadIdx.x;
    if (idx >= Bc * NN) return;
    int b = idx / NN, xy = idx % NN;
    float a = 0.f;
    for (int c = 0; c < NCc; ++c) a += prob[b * NCc + c] * att[(size_t)c * NN + xy];
    K16[idx] = (_Float16)a;
}

// ---------------------------------------------------------------------------
// Rollout: K (360x360 f16, dense, rows 720B = 45*16B so v8h reads stay
// 16B-aligned) resident in LDS.  The global->LDS stage is done by the Tensor
// Data Mover when available (one 2D descriptor issued by wave 0, completion
// via s_wait_tensorcnt); manual cooperative copy otherwise.  One block per
// (b,dir); serial matvec iterations with a double-buffered carry vector.
// ---------------------------------------------------------------------------
#if defined(__has_builtin)
#  if __has_builtin(__builtin_amdgcn_tensor_load_to_lds) && \
      __has_builtin(__builtin_amdgcn_s_wait_tensorcnt)
#    define KNF_HAVE_TDM 1
#  endif
#endif

__global__ void rollout_kernel(const _Float16* __restrict__ K16,
                               const float* __restrict__ z32,
                               float* __restrict__ lb32,
                               _Float16* __restrict__ lb16,
                               _Float16* __restrict__ la16) {
    extern __shared__ char smem[];
    _Float16* Kl = (_Float16*)smem;                 // dynamic-LDS base (offset 0)
    const int NN = Nc * Nc;
    float* v0 = (float*)(Kl + NN);
    float* v1 = v0 + Nc;
    const int b = blockIdx.x >> 1;
    const int dir = blockIdx.x & 1;

#ifdef KNF_HAVE_TDM
    if ((threadIdx.x >> 5) == 0) {
        // D# for a 2D 360x360 tile of 2-byte elements, row stride 360,
        // destination = LDS offset 0 (this kernel's dynamic LDS base).
        uint64_t ga = (uint64_t)(uintptr_t)(K16 + (size_t)b * NN);
        u32x4 g0 = { 1u,                                  // count=1 valid D#
                     0u,                                  // lds_addr = 0
                     (unsigned)ga,                        // global_addr[31:0]
                     (((unsigned)(ga >> 32)) & 0x01FFFFFFu) | 0x80000000u }; // addr[56:32] | type=2
        i32x8 g1 = { (int)0x00010000u,                    // data_size = 2 bytes
                     (int)(((unsigned)Nc) << 16),         // tensor_dim0 lo16
                     (int)(((unsigned)Nc) << 16),         // tensor_dim1 lo16 (hi=0)
                     (int)(((unsigned)Nc) << 16),         // tile_dim0
                     (int)Nc,                             // tile_dim1 (tile_dim2=0)
                     (int)Nc,                             // tensor_dim0_stride lo32
                     0, 0 };                              // stride hi / dim1_stride
        i32x4 gz = { 0, 0, 0, 0 };
#if __clang_major__ >= 23
        i32x8 gz8 = { 0, 0, 0, 0, 0, 0, 0, 0 };
        __builtin_amdgcn_tensor_load_to_lds(g0, g1, gz, gz, gz8, 0);
#else
        __builtin_amdgcn_tensor_load_to_lds(g0, g1, gz, gz, 0);
#endif
        __builtin_amdgcn_s_wait_tensorcnt(0);
    }
#else
    for (int i = threadIdx.x; i < NN; i += blockDim.x)
        Kl[i] = K16[(size_t)b * NN + i];
#endif

    const int row = threadIdx.x;
    if (row < Nc) v0[row] = z32[((size_t)b * Nc + row) * Tc + (dir ? (Tc - 1) : 0)];
    __syncthreads();

    const int iters = dir ? TYc : (Tc - 1);
    for (int it = 0; it < iters; ++it) {
        const float* vin = (it & 1) ? v1 : v0;
        float* vout      = (it & 1) ? v0 : v1;
        if (row < Nc) {
            float acc = 0.f;
            const _Float16* kr = Kl + row * Nc;
#pragma unroll 5
            for (int l8 = 0; l8 < Nc / 8; ++l8) {
                v8h kk = *(const v8h*)(kr + l8 * 8);
                const float* vv = vin + l8 * 8;
                acc += (float)kk[0] * vv[0] + (float)kk[1] * vv[1]
                     + (float)kk[2] * vv[2] + (float)kk[3] * vv[3]
                     + (float)kk[4] * vv[4] + (float)kk[5] * vv[5]
                     + (float)kk[6] * vv[6] + (float)kk[7] * vv[7];
            }
            vout[row] = acc;
            size_t oi = ((size_t)b * Nc + row) * iters + it;
            if (dir == 0) { lb32[oi] = acc; lb16[oi] = (_Float16)acc; }
            else          { la16[oi] = (_Float16)acc; }
        }
        __syncthreads();
    }
}

// MSE with index remap.  mode 0: ref layout (B,RT,N); mode 1: ref (B,N,RT).
__global__ void mse_kernel(const float* __restrict__ pred, const float* __restrict__ ref,
                           int L, int RT, int tshift, int mode, float scale,
                           float* __restrict__ outslot) {
    __shared__ float red[256];
    const size_t total = (size_t)Bc * Nc * L;
    float s = 0.f;
    for (size_t i = (size_t)blockIdx.x * blockDim.x + threadIdx.x; i < total;
         i += (size_t)gridDim.x * blockDim.x) {
        int t = (int)(i % L);
        size_t r_ = i / L;
        int n = (int)(r_ % Nc);
        int b = (int)(r_ / Nc);
        size_t ridx = (mode == 0) ? (((size_t)b * RT + t + tshift) * Nc + n)
                                  : (((size_t)b * Nc + n) * RT + t + tshift);
        float d = pred[i] - ref[ridx];
        s += d * d;
    }
    red[threadIdx.x] = s;
    __syncthreads();
    for (int off = 128; off > 0; off >>= 1) {
        if ((int)threadIdx.x < off) red[threadIdx.x] += red[threadIdx.x + off];
        __syncthreads();
    }
    if (threadIdx.x == 0) atomicAdd(outslot, red[0] * scale);
}

__global__ void zero4_kernel(float* p) {
    if (threadIdx.x < 4) p[threadIdx.x] = 0.f;
}

// ---------------------------------------------------------------------------
extern "C" void kernel_launch(void* const* d_in, const int* in_sizes, int n_in,
                              void* d_out, int out_size, void* d_ws, size_t ws_size,
                              hipStream_t stream) {
    (void)in_sizes; (void)n_in; (void)out_size; (void)ws_size;
    const float* x        = (const float*)d_in[0];
    const float* y        = (const float*)d_in[1];
    const int*   labels   = (const int*)d_in[2];
    const float* conv_w0  = (const float*)d_in[3];
    const float* conv_b0  = (const float*)d_in[4];
    const float* conv_w1  = (const float*)d_in[5];
    const float* conv_b1  = (const float*)d_in[6];
    const float* conv_w2  = (const float*)d_in[7];
    const float* conv_b2  = (const float*)d_in[8];
    const float* proj_w   = (const float*)d_in[9];
    const float* proj_b   = (const float*)d_in[10];
    const float* convt_w0 = (const float*)d_in[11];
    const float* convt_b0 = (const float*)d_in[12];
    const float* convt_w1 = (const float*)d_in[13];
    const float* convt_b1 = (const float*)d_in[14];
    const float* convt_w2 = (const float*)d_in[15];
    const float* convt_b2 = (const float*)d_in[16];
    const float* projt_w  = (const float*)d_in[17];
    const float* projt_b  = (const float*)d_in[18];
    const float* attention= (const float*)d_in[19];
    const float* cls_w0   = (const float*)d_in[20];
    const float* cls_b0   = (const float*)d_in[21];
    const float* cls_w1   = (const float*)d_in[22];
    const float* cls_b1   = (const float*)d_in[23];
    const float* dense_w  = (const float*)d_in[24];
    const float* dense_b  = (const float*)d_in[25];
    float* out = (float*)d_out;

    // --- workspace carve (~18 MB) ---
    char* ws = (char*)d_ws;
    size_t off = 0;
    auto carve = [&](size_t bytes) -> void* {
        void* p = ws + off;
        off = (off + bytes + 255) & ~(size_t)255;
        return p;
    };
    const int KC64 = 6, KCc0 = (Nc * 3 + 31) / 32 /*34*/, KCc1 = (CHc * 3 + 31) / 32 /*12*/;
    const int tot_e  = 4 * KC64 * 512;
    const int tot_c0 = 8 * KCc0 * 512;
    const int tot_c1 = 8 * KCc1 * 512;
    _Float16* ape1 = (_Float16*)carve((size_t)tot_e * 2);
    _Float16* ape2 = (_Float16*)carve((size_t)tot_e * 2);
    _Float16* apd1 = (_Float16*)carve((size_t)tot_e * 2);
    _Float16* apd2 = (_Float16*)carve((size_t)tot_e * 2);
    _Float16* apc0 = (_Float16*)carve((size_t)tot_c0 * 2);
    _Float16* apc1 = (_Float16*)carve((size_t)tot_c1 * 2);
    _Float16* z16  = (_Float16*)carve((size_t)Bc * Nc * Tc * 2);
    float*    z32  = (float*)  carve((size_t)Bc * Nc * Tc * 4);
    _Float16* ch1  = (_Float16*)carve((size_t)Bc * CHc * Tc * 2);
    _Float16* ch2  = (_Float16*)carve((size_t)Bc * CHc * Tc * 2);
    float*    pooled=(float*)  carve((size_t)Bc * CHc * 4);
    float*    prob = (float*)  carve((size_t)Bc * NCc * 4);
    _Float16* K16  = (_Float16*)carve((size_t)Bc * Nc * Nc * 2);
    float*    lb32 = (float*)  carve((size_t)Bc * Nc * (Tc - 1) * 4);
    _Float16* lb16 = (_Float16*)carve((size_t)Bc * Nc * (Tc - 1) * 2);
    _Float16* la16 = (_Float16*)carve((size_t)Bc * Nc * TYc * 2);

    // --- weight fragment packing ---
    pack_a_kernel<<<(tot_e  + 255) / 256, 256, 0, stream>>>(conv_w1,  ape1, Hc,  Hc, KC64, 0, tot_e);
    pack_a_kernel<<<(tot_e  + 255) / 256, 256, 0, stream>>>(conv_w2,  ape2, Hc,  Hc, KC64, 0, tot_e);
    pack_a_kernel<<<(tot_e  + 255) / 256, 256, 0, stream>>>(convt_w1, apd1, Hc,  Hc, KC64, 1, tot_e);
    pack_a_kernel<<<(tot_e  + 255) / 256, 256, 0, stream>>>(convt_w2, apd2, Hc,  Hc, KC64, 1, tot_e);
    pack_a_kernel<<<(tot_c0 + 255) / 256, 256, 0, stream>>>(cls_w0,   apc0, CHc, Nc,  KCc0, 0, tot_c0);
    pack_a_kernel<<<(tot_c1 + 255) / 256, 256, 0, stream>>>(cls_w1,   apc1, CHc, CHc, KCc1, 0, tot_c1);

    zero4_kernel<<<1, 32, 0, stream>>>(out + O_SCAL);

    // --- encode (fused, LDS-resident) ---
    const size_t smemE = (size_t)2 * 64 * TROWS * 2 + (size_t)Tc * 4;   // ~100.9 KB
    encode_fused_kernel<<<Bc * Nc, 128, smemE, stream>>>(
        x, conv_w0, conv_b0, ape1, conv_b1, ape2, conv_b2, proj_w, proj_b, z32, z16);

    // --- classifier ---
    const int ntilesT = (Tc + 15) / 16;
    const int nwaves  = Bc * ntilesT;                 // 192 waves
    conv1d_wmma_global_kernel<<<(nwaves * 32 + 255) / 256, 256, 0, stream>>>(
        z16, apc0, cls_b0, ch1, Nc, Tc, 1);
    conv1d_wmma_global_kernel<<<(nwaves * 32 + 255) / 256, 256, 0, stream>>>(
        ch1, apc1, cls_b1, ch2, CHc, Tc, 0);
    meanpool_kernel<<<(Bc * CHc + 255) / 256, 256, 0, stream>>>(ch2, pooled);
    head_kernel<<<1, 192, 0, stream>>>(pooled, dense_w, dense_b, labels,
                                       out + O_SCORES, prob, out + O_SCAL + 4);

    // --- Koopman operator + rollouts ---
    kbuild_kernel<<<(Bc * Nc * Nc + 255) / 256, 256, 0, stream>>>(prob, attention, K16);
    const size_t smemR = (size_t)Nc * Nc * 2 + (size_t)2 * Nc * 4;   // ~262 KB
    rollout_kernel<<<2 * Bc, 384, smemR, stream>>>(K16, z32, lb32, lb16, la16);

    // --- decode (fused) x3 ---
    const size_t smemD = (size_t)2 * 64 * TROWS * 2;                 // ~99.3 KB
    decode_fused_kernel<<<Bc * Nc, 128, smemD, stream>>>(
        z16, convt_w0, convt_b0, apd1, convt_b1, apd2, convt_b2,
        projt_w, projt_b, out + O_XREC, Tc);
    decode_fused_kernel<<<Bc * Nc, 128, smemD, stream>>>(
        lb16, convt_w0, convt_b0, apd1, convt_b1, apd2, convt_b2,
        projt_w, projt_b, out + O_LBREC, Tc - 1);
    decode_fused_kernel<<<Bc * Nc, 128, smemD, stream>>>(
        la16, convt_w0, convt_b0, apd1, convt_b1, apd2, convt_b2,
        projt_w, projt_b, out + O_LAREC, TYc);

    // --- losses ---
    mse_kernel<<<256, 256, 0, stream>>>(out + O_XREC,  x,   Tc,     Tc, 0, 0,
                                        1.0f / (float)SZ_XREC,  out + O_SCAL + 0);
    mse_kernel<<<256, 256, 0, stream>>>(out + O_LBREC, x,   Tc - 1, Tc, 1, 0,
                                        1.0f / (float)SZ_LBREC, out + O_SCAL + 1);
    mse_kernel<<<256, 256, 0, stream>>>(out + O_LAREC, y,   TYc,    TYc, 0, 0,
                                        1.0f / (float)SZ_LAREC, out + O_SCAL + 2);
    mse_kernel<<<256, 256, 0, stream>>>(lb32,          z32, Tc - 1, Tc, 1, 1,
                                        1.0f / (float)SZ_LBREC, out + O_SCAL + 3);
}
